// GraphMHSA_58342835749135
// MI455X (gfx1250) — matrix operand order repaired and checked by reading, not compiled, not run on hardware
//
#include <hip/hip_runtime.h>

// ---------------------------------------------------------------------------
// Fused MHSA forward for MI455X (gfx1250, wave32, WMMA bf16 16x16x32).
// cvt(fp32->bf16, weights transposed) -> QKV gemm -> flash attn -> proj gemm
// A fragments load as contiguous 16-byte ops, B fragments as strided bf16 loads; LDS double-buffered;
// accumulators in named registers to avoid allocator copies around v_wmma.
// ---------------------------------------------------------------------------

typedef __attribute__((ext_vector_type(16))) __bf16 v16bf;
typedef __attribute__((ext_vector_type(8)))  __bf16 v8bf;
typedef __attribute__((ext_vector_type(8)))  float  v8f;

#define D_MODEL 1024
#define N_HEADS 16
#define D_HEAD  64
#define BATCH   4
#define SEQ     1024
#define M_TOT   (BATCH * SEQ)      // 4096
#define N_QKV   (3 * D_MODEL)      // 3072

__device__ __forceinline__ v8f wmma_bf16(v16bf a, v16bf b, v8f c) {
  return __builtin_amdgcn_wmma_f32_16x16x32_bf16(false, a, false, b, (short)0,
                                                 c, false, false);
}

// A operand (16x32 MxK) from row-major tile with rowStride=32.
// lanes 0-15: row M=lane, K={0..7,16..23}; lanes 16-31: row M=lane-16,
// K={8..15,24..31}  ->  two contiguous 16B chunks per lane.
__device__ __forceinline__ v16bf a_frag32(const __bf16* tile, int mbase,
                                          int lane) {
  const int kb = (lane & 16) ? 8 : 0;
  const __bf16* p = tile + (mbase + (lane & 15)) * 32;
  const v8bf lo = *(const v8bf*)(p + kb);
  const v8bf hi = *(const v8bf*)(p + 16 + kb);
  v16bf a;
#pragma unroll
  for (int e = 0; e < 8; ++e) { a[e] = lo[e]; a[e + 8] = hi[e]; }
  return a;
}

// B operand (32x16 KxN) from K-major tile: lane%16 = n column, elements =
// K={0..7,16..23} (lanes 0-15) / K={8..15,24..31} (lanes 16-31).
__device__ __forceinline__ v16bf b_frag(const __bf16* tile, int rowStride,
                                        int n0, int lane) {
  const int kb = (lane & 16) ? 8 : 0;
  const __bf16* p = tile + n0 + (lane & 15);
  v16bf b;
#pragma unroll
  for (int e = 0; e < 8; ++e) {
    b[e] = p[(kb + e) * rowStride];
    b[e + 8] = p[(16 + kb + e) * rowStride];
  }
  return b;
}

// ---------------------------------------------------------------------------
// fp32 -> bf16 elementwise convert (grid-stride)
// ---------------------------------------------------------------------------
__global__ void mhsa_cvt_bf16(const float* __restrict__ in,
                              __bf16* __restrict__ out, int n) {
  int i = blockIdx.x * blockDim.x + threadIdx.x;
  const int stride = gridDim.x * blockDim.x;
  for (; i < n; i += stride) out[i] = (__bf16)in[i];
}

// fp32 [N][K] -> bf16 [K][N] transpose+convert (one-off, weights only)
__global__ void mhsa_cvt_t_bf16(const float* __restrict__ in,
                                __bf16* __restrict__ out, int N, int K) {
  int i = blockIdx.x * blockDim.x + threadIdx.x;
  const int stride = gridDim.x * blockDim.x;
  const int total = N * K;
  for (; i < total; i += stride) {
    const int k = i / N;
    const int n = i - k * N;
    out[i] = (__bf16)in[n * K + k];
  }
}

// ---------------------------------------------------------------------------
// QKV projection: qkv[m,n] = sum_k x[m,k] * Wt[k,n] + b[n]
// Block = 256 threads (8 waves), 128x128 tile; wave = 32x64 (8 WMMAs/K-step).
// LDS double-buffered, one barrier per K-step. Scatters into:
//   Q: [B,H,T,dh]   K: [B,H,dh,T] (transposed for attention)   V: [B,H,T,dh]
// ---------------------------------------------------------------------------
__global__ __launch_bounds__(256)
void mhsa_qkv_gemm(const __bf16* __restrict__ X,   // [4096][1024]
                   const __bf16* __restrict__ Wt,  // [1024][3072] k-major
                   const float*  __restrict__ bias,
                   __bf16* __restrict__ Q, __bf16* __restrict__ K,
                   __bf16* __restrict__ V) {
  __shared__ __align__(16) __bf16 xs[2][128 * 32];  // [m][k]
  __shared__ __align__(16) __bf16 wt[2][32 * 128];  // [k][n]

  const int tid  = threadIdx.x;
  const int lane = tid & 31;
  const int wv   = tid >> 5;
  const int m0   = blockIdx.y * 128;
  const int n0   = blockIdx.x * 128;
  const int wm   = (wv & 3) * 32;
  const int wn   = (wv >> 2) * 64;

  const int xrow = tid >> 1, xseg = (tid & 1) * 16;  // 2 uint4 of X / thread
  const int krow = tid >> 3, nseg = (tid & 7) * 16;  // 2 uint4 of Wt / thread

  v8f c00 = {}, c01 = {}, c02 = {}, c03 = {};
  v8f c10 = {}, c11 = {}, c12 = {}, c13 = {};

  // prologue: chunk 0 -> buffer 0
  {
    uint4 rx0 = *(const uint4*)&X[(m0 + xrow) * D_MODEL + xseg];
    uint4 rx1 = *(const uint4*)&X[(m0 + xrow) * D_MODEL + xseg + 8];
    uint4 rw0 = *(const uint4*)&Wt[krow * N_QKV + n0 + nseg];
    uint4 rw1 = *(const uint4*)&Wt[krow * N_QKV + n0 + nseg + 8];
    *(uint4*)&xs[0][xrow * 32 + xseg]     = rx0;
    *(uint4*)&xs[0][xrow * 32 + xseg + 8] = rx1;
    *(uint4*)&wt[0][krow * 128 + nseg]     = rw0;
    *(uint4*)&wt[0][krow * 128 + nseg + 8] = rw1;
  }

  for (int it = 0; it < D_MODEL / 32; ++it) {
    const int cur = it & 1, nxt = cur ^ 1;
    const int kc = (it + 1) * 32;
    const bool has = kc < D_MODEL;
    uint4 rx0 = {}, rx1 = {}, rw0 = {}, rw1 = {};
    if (has) {  // issue next-chunk global loads (wait lands after WMMAs)
      rx0 = *(const uint4*)&X[(m0 + xrow) * D_MODEL + kc + xseg];
      rx1 = *(const uint4*)&X[(m0 + xrow) * D_MODEL + kc + xseg + 8];
      rw0 = *(const uint4*)&Wt[(kc + krow) * N_QKV + n0 + nseg];
      rw1 = *(const uint4*)&Wt[(kc + krow) * N_QKV + n0 + nseg + 8];
    }
    __syncthreads();
    const v16bf a0 = a_frag32(xs[cur], wm, lane);
    const v16bf a1 = a_frag32(xs[cur], wm + 16, lane);
    {
      const v16bf b = b_frag(wt[cur], 128, wn, lane);
      c00 = wmma_bf16(a0, b, c00);
      c10 = wmma_bf16(a1, b, c10);
    }
    {
      const v16bf b = b_frag(wt[cur], 128, wn + 16, lane);
      c01 = wmma_bf16(a0, b, c01);
      c11 = wmma_bf16(a1, b, c11);
    }
    {
      const v16bf b = b_frag(wt[cur], 128, wn + 32, lane);
      c02 = wmma_bf16(a0, b, c02);
      c12 = wmma_bf16(a1, b, c12);
    }
    {
      const v16bf b = b_frag(wt[cur], 128, wn + 48, lane);
      c03 = wmma_bf16(a0, b, c03);
      c13 = wmma_bf16(a1, b, c13);
    }
    if (has) {
      *(uint4*)&xs[nxt][xrow * 32 + xseg]     = rx0;
      *(uint4*)&xs[nxt][xrow * 32 + xseg + 8] = rx1;
      *(uint4*)&wt[nxt][krow * 128 + nseg]     = rw0;
      *(uint4*)&wt[nxt][krow * 128 + nseg + 8] = rw1;
    }
  }

  // Epilogue: bias add + scatter into Q / K(transposed) / V
  const int colL = lane & 15;
  const int rowH = (lane >> 4) * 8;
#pragma unroll
  for (int nt = 0; nt < 4; ++nt) {
    const v8f am0 = (nt == 0) ? c00 : (nt == 1) ? c01 : (nt == 2) ? c02 : c03;
    const v8f am1 = (nt == 0) ? c10 : (nt == 1) ? c11 : (nt == 2) ? c12 : c13;
    const int n     = n0 + wn + nt * 16 + colL;
    const int which = n >> 10;                 // 0=Q 1=K 2=V
    const int rem   = n & (D_MODEL - 1);
    const int head  = rem >> 6;
    const int dh    = rem & (D_HEAD - 1);
    const float bv  = bias[n];
#pragma unroll
    for (int mi = 0; mi < 2; ++mi) {
#pragma unroll
      for (int r = 0; r < 8; ++r) {
        const int m  = m0 + wm + mi * 16 + rowH + r;
        const int bb = m >> 10;
        const int tt = m & (SEQ - 1);
        const float val = (mi ? am1[r] : am0[r]) + bv;
        if (which == 1) {  // K: [B,H,dh,T]
          K[((size_t)(bb * N_HEADS + head) * D_HEAD + dh) * SEQ + tt] =
              (__bf16)val;
        } else {
          __bf16* dst = (which == 0) ? Q : V;  // [B,H,T,dh]
          dst[((size_t)(bb * N_HEADS + head) * SEQ + tt) * D_HEAD + dh] =
              (__bf16)val;
        }
      }
    }
  }
}

// ---------------------------------------------------------------------------
// Flash attention with additive bias.
// grid = (T/128, B*H); block = 256 (8 waves); each wave owns a 16-row Q tile.
// K arrives d-major (kT[d][key]); QK^T B-fragments are strided per-element LDS loads.
// Output: bf16 [B, T, H*dh].
// ---------------------------------------------------------------------------
__global__ __launch_bounds__(256)
void mhsa_attn(const __bf16* __restrict__ Q,  // [B,H,T,dh]
               const __bf16* __restrict__ Kt, // [B,H,dh,T]
               const __bf16* __restrict__ V,  // [B,H,T,dh]
               const float* __restrict__ bias,
               __bf16* __restrict__ O) {
  __shared__ __align__(16) __bf16 kT[2][64 * 32];   // [d][key]
  __shared__ __align__(16) __bf16 vl[2][32 * 64];   // [key][dh]
  __shared__ __align__(16) __bf16 plds[8][16 * 32]; // per-wave P scratch

  const int tid  = threadIdx.x;
  const int lane = tid & 31;
  const int wv   = tid >> 5;
  const int bh   = blockIdx.y;
  const int q0   = blockIdx.x * 128 + wv * 16;

  const __bf16* Qh = Q  + (size_t)bh * SEQ * D_HEAD;
  const __bf16* Kh = Kt + (size_t)bh * D_HEAD * SEQ;
  const __bf16* Vh = V  + (size_t)bh * SEQ * D_HEAD;
  const float*  Bh = bias + (size_t)bh * SEQ * SEQ;

  // Q fragments (rows q0..q0+15; dh split into two K=32 fragments)
  v16bf aq0, aq1;
  {
    const int kb = (lane & 16) ? 8 : 0;
    const __bf16* p = &Qh[(q0 + (lane & 15)) * D_HEAD];
    const v8bf l0 = *(const v8bf*)(p + kb);
    const v8bf h0 = *(const v8bf*)(p + 16 + kb);
    const v8bf l1 = *(const v8bf*)(p + 32 + kb);
    const v8bf h1 = *(const v8bf*)(p + 48 + kb);
#pragma unroll
    for (int e = 0; e < 8; ++e) {
      aq0[e] = l0[e]; aq0[e + 8] = h0[e];
      aq1[e] = l1[e]; aq1[e + 8] = h1[e];
    }
  }

  v8f o0 = {}, o1 = {}, o2 = {}, o3 = {};
  float mi[8], li[8];
#pragma unroll
  for (int r = 0; r < 8; ++r) { mi[r] = -3.0e38f; li[r] = 0.0f; }

  const float sc   = 0.125f;        // 1/sqrt(64)
  const int   drow = tid >> 2, kseg = (tid & 3) * 8;  // kT copy: 1 uint4
  const int   vrow = tid >> 3, vseg = (tid & 7) * 8;  // vl copy: 1 uint4
  const int   colL = lane & 15;
  const int   rowH = (lane >> 4) * 8;

  // prologue: key chunk 0 -> buffer 0
  *(uint4*)&kT[0][drow * 32 + kseg] = *(const uint4*)&Kh[drow * SEQ + kseg];
  *(uint4*)&vl[0][vrow * 64 + vseg] = *(const uint4*)&Vh[vrow * D_HEAD + vseg];

  for (int it = 0; it < SEQ / 32; ++it) {
    const int cur = it & 1, nxt = cur ^ 1;
    const int kc = it * 32;
    const bool has = kc + 32 < SEQ;
    uint4 rk = {}, rv = {};
    if (has) {
      rk = *(const uint4*)&Kh[drow * SEQ + kc + 32 + kseg];
      rv = *(const uint4*)&Vh[(kc + 32 + vrow) * D_HEAD + vseg];
      __builtin_prefetch(&Bh[(size_t)(q0 + rowH) * SEQ + kc + 32], 0, 0);
    }
    __syncthreads();

    // S = Q.K^T : B operand lane = key column of kT, elements = d rows (stride 32)
    v8f s0 = {}, s1 = {};
    s0 = wmma_bf16(aq0, b_frag(kT[cur], 32, 0, lane), s0);
    s0 = wmma_bf16(aq1, b_frag(kT[cur] + 32 * 32, 32, 0, lane), s0);
    s1 = wmma_bf16(aq0, b_frag(kT[cur], 32, 16, lane), s1);
    s1 = wmma_bf16(aq1, b_frag(kT[cur] + 32 * 32, 32, 16, lane), s1);

    // scale + additive bias (dominant HBM stream)
#pragma unroll
    for (int r = 0; r < 8; ++r) {
      const float* bp = &Bh[(size_t)(q0 + rowH + r) * SEQ + kc];
      s0[r] = s0[r] * sc + bp[colL];
      s1[r] = s1[r] * sc + bp[16 + colL];
    }

    // online softmax (row reductions over the 16-lane half)
    float p0[8], p1[8];
#pragma unroll
    for (int r = 0; r < 8; ++r) {
      float mx = fmaxf(s0[r], s1[r]);
#pragma unroll
      for (int off = 8; off >= 1; off >>= 1)
        mx = fmaxf(mx, __shfl_xor(mx, off, 32));
      const float mnew = fmaxf(mi[r], mx);
      const float corr = __expf(mi[r] - mnew);
      mi[r] = mnew;
      p0[r] = __expf(s0[r] - mnew);
      p1[r] = __expf(s1[r] - mnew);
      float rs = p0[r] + p1[r];
#pragma unroll
      for (int off = 8; off >= 1; off >>= 1)
        rs += __shfl_xor(rs, off, 32);
      li[r] = li[r] * corr + rs;
      o0[r] *= corr; o1[r] *= corr; o2[r] *= corr; o3[r] *= corr;
    }

    // Re-layout P (C layout -> A layout) via per-wave LDS scratch
    __bf16* pw = plds[wv];
#pragma unroll
    for (int r = 0; r < 8; ++r) {
      pw[(rowH + r) * 32 + colL]      = (__bf16)p0[r];
      pw[(rowH + r) * 32 + 16 + colL] = (__bf16)p1[r];
    }
    const v16bf ap = a_frag32(pw, 0, lane);

    // O += P.V : B operand lane = dh column of vl, elements = key rows (stride 64)
    o0 = wmma_bf16(ap, b_frag(vl[cur], 64, 0,  lane), o0);
    o1 = wmma_bf16(ap, b_frag(vl[cur], 64, 16, lane), o1);
    o2 = wmma_bf16(ap, b_frag(vl[cur], 64, 32, lane), o2);
    o3 = wmma_bf16(ap, b_frag(vl[cur], 64, 48, lane), o3);

    if (has) {
      *(uint4*)&kT[nxt][drow * 32 + kseg] = rk;
      *(uint4*)&vl[nxt][vrow * 64 + vseg] = rv;
    }
  }

  // finalize: divide by l, store bf16 into [B, T, H*dh]
  const int h  = bh & (N_HEADS - 1);
  const int bb = bh >> 4;
#pragma unroll
  for (int r = 0; r < 8; ++r) {
    const float inv = 1.0f / li[r];
    const int t = q0 + rowH + r;
    __bf16* dst = &O[((size_t)(bb * SEQ + t)) * D_MODEL + h * D_HEAD];
    dst[colL]      = (__bf16)(o0[r] * inv);
    dst[16 + colL] = (__bf16)(o1[r] * inv);
    dst[32 + colL] = (__bf16)(o2[r] * inv);
    dst[48 + colL] = (__bf16)(o3[r] * inv);
  }
}

// ---------------------------------------------------------------------------
// Output projection: out[m,n] = sum_k A[m,k] * Wt[k,n] + b[n]  (fp32 out)
// Same 128x128 / 32x64-per-wave double-buffered structure.
// ---------------------------------------------------------------------------
__global__ __launch_bounds__(256)
void mhsa_proj_gemm(const __bf16* __restrict__ A,   // [4096][1024]
                    const __bf16* __restrict__ Wt,  // [1024][1024] k-major
                    const float*  __restrict__ bias,
                    float* __restrict__ out) {
  __shared__ __align__(16) __bf16 xs[2][128 * 32];
  __shared__ __align__(16) __bf16 wt[2][32 * 128];

  const int tid  = threadIdx.x;
  const int lane = tid & 31;
  const int wv   = tid >> 5;
  const int m0   = blockIdx.y * 128;
  const int n0   = blockIdx.x * 128;
  const int wm   = (wv & 3) * 32;
  const int wn   = (wv >> 2) * 64;

  const int xrow = tid >> 1, xseg = (tid & 1) * 16;
  const int krow = tid >> 3, nseg = (tid & 7) * 16;

  v8f c00 = {}, c01 = {}, c02 = {}, c03 = {};
  v8f c10 = {}, c11 = {}, c12 = {}, c13 = {};

  {
    uint4 rx0 = *(const uint4*)&A[(m0 + xrow) * D_MODEL + xseg];
    uint4 rx1 = *(const uint4*)&A[(m0 + xrow) * D_MODEL + xseg + 8];
    uint4 rw0 = *(const uint4*)&Wt[krow * D_MODEL + n0 + nseg];
    uint4 rw1 = *(const uint4*)&Wt[krow * D_MODEL + n0 + nseg + 8];
    *(uint4*)&xs[0][xrow * 32 + xseg]     = rx0;
    *(uint4*)&xs[0][xrow * 32 + xseg + 8] = rx1;
    *(uint4*)&wt[0][krow * 128 + nseg]     = rw0;
    *(uint4*)&wt[0][krow * 128 + nseg + 8] = rw1;
  }

  for (int it = 0; it < D_MODEL / 32; ++it) {
    const int cur = it & 1, nxt = cur ^ 1;
    const int kc = (it + 1) * 32;
    const bool has = kc < D_MODEL;
    uint4 rx0 = {}, rx1 = {}, rw0 = {}, rw1 = {};
    if (has) {
      rx0 = *(const uint4*)&A[(m0 + xrow) * D_MODEL + kc + xseg];
      rx1 = *(const uint4*)&A[(m0 + xrow) * D_MODEL + kc + xseg + 8];
      rw0 = *(const uint4*)&Wt[(kc + krow) * D_MODEL + n0 + nseg];
      rw1 = *(const uint4*)&Wt[(kc + krow) * D_MODEL + n0 + nseg + 8];
    }
    __syncthreads();
    const v16bf a0 = a_frag32(xs[cur], wm, lane);
    const v16bf a1 = a_frag32(xs[cur], wm + 16, lane);
    {
      const v16bf b = b_frag(wt[cur], 128, wn, lane);
      c00 = wmma_bf16(a0, b, c00);
      c10 = wmma_bf16(a1, b, c10);
    }
    {
      const v16bf b = b_frag(wt[cur], 128, wn + 16, lane);
      c01 = wmma_bf16(a0, b, c01);
      c11 = wmma_bf16(a1, b, c11);
    }
    {
      const v16bf b = b_frag(wt[cur], 128, wn + 32, lane);
      c02 = wmma_bf16(a0, b, c02);
      c12 = wmma_bf16(a1, b, c12);
    }
    {
      const v16bf b = b_frag(wt[cur], 128, wn + 48, lane);
      c03 = wmma_bf16(a0, b, c03);
      c13 = wmma_bf16(a1, b, c13);
    }
    if (has) {
      *(uint4*)&xs[nxt][xrow * 32 + xseg]     = rx0;
      *(uint4*)&xs[nxt][xrow * 32 + xseg + 8] = rx1;
      *(uint4*)&wt[nxt][krow * 128 + nseg]     = rw0;
      *(uint4*)&wt[nxt][krow * 128 + nseg + 8] = rw1;
    }
  }

  const int colL = lane & 15;
  const int rowH = (lane >> 4) * 8;
#pragma unroll
  for (int nt = 0; nt < 4; ++nt) {
    const v8f am0 = (nt == 0) ? c00 : (nt == 1) ? c01 : (nt == 2) ? c02 : c03;
    const v8f am1 = (nt == 0) ? c10 : (nt == 1) ? c11 : (nt == 2) ? c12 : c13;
    const int n = n0 + wn + nt * 16 + colL;
    const float bv = bias[n];
#pragma unroll
    for (int mi = 0; mi < 2; ++mi) {
#pragma unroll
      for (int r = 0; r < 8; ++r) {
        const int m = m0 + wm + mi * 16 + rowH + r;
        out[(size_t)m * D_MODEL + n] = (mi ? am1[r] : am0[r]) + bv;
      }
    }
  }
}

// ---------------------------------------------------------------------------
// Host launcher
// ---------------------------------------------------------------------------
extern "C" void kernel_launch(void* const* d_in, const int* in_sizes, int n_in,
                              void* d_out, int out_size, void* d_ws,
                              size_t ws_size, hipStream_t stream) {
  (void)in_sizes; (void)n_in; (void)out_size; (void)ws_size;

  const float* x         = (const float*)d_in[0];
  const float* attn_bias = (const float*)d_in[1];
  const float* w_qkv     = (const float*)d_in[2];
  const float* b_qkv     = (const float*)d_in[3];
  const float* w_proj    = (const float*)d_in[4];
  const float* b_proj    = (const float*)d_in[5];
  float*       out       = (float*)d_out;

  char* ws = (char*)d_ws;
  const size_t SZ_X    = (size_t)M_TOT * D_MODEL * 2;                 // 8 MB
  const size_t SZ_WQ   = (size_t)N_QKV * D_MODEL * 2;                 // 6 MB
  const size_t SZ_WP   = (size_t)D_MODEL * D_MODEL * 2;               // 2 MB
  const size_t SZ_HEAD = (size_t)BATCH * N_HEADS * SEQ * D_HEAD * 2;  // 8 MB

  __bf16* x_bf     = (__bf16*)(ws);
  __bf16* wqkv_t   = (__bf16*)(ws + SZ_X);
  __bf16* wproj_t  = (__bf16*)(ws + SZ_X + SZ_WQ);
  __bf16* Qb       = (__bf16*)(ws + SZ_X + SZ_WQ + SZ_WP);
  __bf16* Kb       = (__bf16*)(ws + SZ_X + SZ_WQ + SZ_WP + SZ_HEAD);
  __bf16* Vb       = (__bf16*)(ws + SZ_X + SZ_WQ + SZ_WP + 2 * SZ_HEAD);
  __bf16* attn_o   = (__bf16*)(ws + SZ_X + SZ_WQ + SZ_WP + 3 * SZ_HEAD);

  // 1) converts (weights transposed to k-major for contiguous B tiles)
  mhsa_cvt_bf16<<<2048, 256, 0, stream>>>(x, x_bf, M_TOT * D_MODEL);
  mhsa_cvt_t_bf16<<<2048, 256, 0, stream>>>(w_qkv, wqkv_t, N_QKV, D_MODEL);
  mhsa_cvt_t_bf16<<<2048, 256, 0, stream>>>(w_proj, wproj_t, D_MODEL, D_MODEL);

  // 2) QKV projection -> Q [B,H,T,dh], K [B,H,dh,T], V [B,H,T,dh]
  mhsa_qkv_gemm<<<dim3(N_QKV / 128, M_TOT / 128), 256, 0, stream>>>(
      x_bf, wqkv_t, b_qkv, Qb, Kb, Vb);

  // 3) Flash attention with bias -> bf16 [B,T,H*dh]
  mhsa_attn<<<dim3(SEQ / 128, BATCH * N_HEADS), 256, 0, stream>>>(
      Qb, Kb, Vb, attn_bias, attn_o);

  // 4) Output projection -> fp32 out
  mhsa_proj_gemm<<<dim3(D_MODEL / 128, M_TOT / 128), 256, 0, stream>>>(
      attn_o, wproj_t, b_proj, out);
}
